// MultiHeadAttention_56152402428358
// MI455X (gfx1250) — matrix operand order repaired and checked by reading, not compiled, not run on hardware
//
#include <hip/hip_runtime.h>
#include <hip/hip_bf16.h>

// ---------------------------------------------------------------------------
// MI455X (gfx1250) multi-head attention with memory mask.
// All GEMMs: v_wmma_f32_16x16x32_bf16, operands pre-converted to bf16,
// tiles moved global->LDS by the Tensor Data Mover (double buffered).
// ---------------------------------------------------------------------------

typedef __attribute__((ext_vector_type(8)))  float  v8f;
typedef __attribute__((ext_vector_type(8)))  __bf16 bf16x8;
typedef __attribute__((ext_vector_type(16))) __bf16 bf16x16;
typedef unsigned int u32x4 __attribute__((ext_vector_type(4)));
typedef int          i32x4 __attribute__((ext_vector_type(4)));
typedef int          i32x8 __attribute__((ext_vector_type(8)));

union FragAB {
  bf16x16 v;
  bf16x8  h[2];
  __bf16  e[16];
};

__device__ __forceinline__ unsigned short f2bf(float f) {
  unsigned u = __builtin_bit_cast(unsigned, f);
  u += 0x7FFFu + ((u >> 16) & 1u);      // round-to-nearest-even
  return (unsigned short)(u >> 16);
}

__device__ __forceinline__ unsigned long long pack4bf(float4 f) {
  return (unsigned long long)f2bf(f.x)
       | ((unsigned long long)f2bf(f.y) << 16)
       | ((unsigned long long)f2bf(f.z) << 32)
       | ((unsigned long long)f2bf(f.w) << 48);
}

#if __has_builtin(__builtin_amdgcn_tensor_load_to_lds) && \
    __has_builtin(__builtin_amdgcn_s_wait_tensorcnt)
#define USE_TDM 1
#else
#define USE_TDM 0
#endif

#if USE_TDM
// Issue one 2D-tile TDM load: tile_w contiguous elements (bf16) per row,
// tile_h rows, row stride in elements.  D# bitfields per CDNA5 ISA 8.3/8.4.
__device__ __forceinline__ void tdm_issue_2d(unsigned lds_off, const void* gptr,
                                             unsigned tile_w, unsigned tile_h,
                                             unsigned long long stride_elems)
{
  unsigned long long ga = (unsigned long long)(size_t)gptr;
  u32x4 g0;
  g0[0] = 1u;                                         // count=1, user descriptor
  g0[1] = lds_off;                                    // LDS byte address
  g0[2] = (unsigned)ga;                               // global_addr[31:0]
  g0[3] = (unsigned)((ga >> 32) & 0x01FFFFFFull)      // global_addr[56:32]
        | (2u << 30);                                 // type=2 ("image")
  i32x8 g1;
  g1[0] = (int)(1u << 16);                            // data_size=1 (2 bytes)
  g1[1] = (int)((tile_w & 0xFFFFu) << 16);            // tensor_dim0[15:0]
  g1[2] = (int)(((tile_w >> 16) & 0xFFFFu)            // tensor_dim0[31:16]
        | ((tile_h & 0xFFFFu) << 16));                // tensor_dim1[15:0]
  g1[3] = (int)(((tile_h >> 16) & 0xFFFFu)            // tensor_dim1[31:16]
        | ((tile_w & 0xFFFFu) << 16));                // tile_dim0
  g1[4] = (int)(tile_h & 0xFFFFu);                    // tile_dim1 (tile_dim2=0)
  g1[5] = (int)(unsigned)(stride_elems & 0xFFFFFFFFull);   // dim0 stride lo
  g1[6] = (int)(unsigned)((stride_elems >> 32) & 0xFFFFull);
  g1[7] = 0;
  i32x4 z4 = {0, 0, 0, 0};
#if __clang_major__ >= 23
  i32x8 z8 = {0, 0, 0, 0, 0, 0, 0, 0};
  __builtin_amdgcn_tensor_load_to_lds(g0, g1, z4, z4, z8, 0);
#else
  __builtin_amdgcn_tensor_load_to_lds(g0, g1, z4, z4, 0);
#endif
}
#endif  // USE_TDM

// ---------------------------------------------------------------------------
// Generic WMMA GEMM:  C = alpha * A[M,K](bf16) * B^T + bias.
// B stored [N,K] bf16, or [K,N] bf16 when TRANSB.
// 128x128 block tile, 256 threads = 8 waves, each wave 64(M)x32(N) via 4x2
// accumulators of v_wmma_f32_16x16x32_bf16; K-step 32; TDM double buffering.
// MODE: 0 = f32 C + bias (+ optional bf16 C2)
//       1 = attention scores: scale + memory mask -> f32 C
//       2 = bf16 C2 only (intermediate results)
// Per-z (b,h) strided views: base += b*s?b + h*s?h with b=z/Hh, h=z%Hh.
// ---------------------------------------------------------------------------
template<int MODE, bool TRANSB>
__global__ __launch_bounds__(256) void wmma_gemm_kernel(
    const unsigned short* __restrict__ A,  long lda, long sAb, long sAh,
    const unsigned short* __restrict__ Bm, long ldb, long sBb, long sBh,
    const float* __restrict__ bias,
    float* __restrict__ C, long ldc, long sCb, long sCh,
    unsigned short* __restrict__ C2, long sC2b, long sC2h,
    int K, int Hh, float alpha, const int* __restrict__ mptr)
{
  const int z  = blockIdx.z;
  const int bb = z / Hh;
  const int hh = z % Hh;
  A  += (size_t)bb * sAb + (size_t)hh * sAh;
  Bm += (size_t)bb * sBb + (size_t)hh * sBh;
  if (MODE != 2) C += (size_t)bb * sCb + (size_t)hh * sCh;
  if (C2)        C2 += (size_t)bb * sC2b + (size_t)hh * sC2h;

  int m_len = 0;
  if (MODE == 1) m_len = *mptr;

  const int bm   = blockIdx.x * 128;
  const int bn   = blockIdx.y * 128;
  const int tid  = threadIdx.x;
  const int wave = tid >> 5;
  const int lane = tid & 31;
  const int wm   = wave & 1;        // 2 waves along M
  const int wn   = wave >> 1;       // 4 waves along N
  const int half = lane >> 4;
  const int l16  = lane & 15;

  // [2] double buffers; tile = 128x32 bf16 = 8 KB each
  __shared__ __align__(32) unsigned short AbufS[2][128 * 32];
  __shared__ __align__(32) unsigned short BbufS[2][128 * 32];

  v8f acc[4][2];
#pragma unroll
  for (int i = 0; i < 4; ++i)
#pragma unroll
    for (int j = 0; j < 2; ++j)
#pragma unroll
      for (int e = 0; e < 8; ++e) acc[i][j][e] = 0.0f;

  const int niter = K >> 5;

#if USE_TDM
  // Prologue: DMA first A/B tiles into buffer 0.
  if (wave == 0) {
    tdm_issue_2d((unsigned)(size_t)&AbufS[0][0], A + (size_t)bm * lda,
                 32u, 128u, (unsigned long long)lda);
    if (!TRANSB)
      tdm_issue_2d((unsigned)(size_t)&BbufS[0][0], Bm + (size_t)bn * ldb,
                   32u, 128u, (unsigned long long)ldb);
    else
      tdm_issue_2d((unsigned)(size_t)&BbufS[0][0], Bm + bn,
                   128u, 32u, (unsigned long long)ldb);
  }
#endif

  for (int it = 0; it < niter; ++it) {
    const int kt = it << 5;
#if USE_TDM
    const int cur = it & 1;
    if (wave == 0) {
      if (it + 1 < niter) {
        const int nb = (it + 1) & 1;
        const int kn = kt + 32;
        tdm_issue_2d((unsigned)(size_t)&AbufS[nb][0],
                     A + (size_t)bm * lda + kn, 32u, 128u,
                     (unsigned long long)lda);
        if (!TRANSB)
          tdm_issue_2d((unsigned)(size_t)&BbufS[nb][0],
                       Bm + (size_t)bn * ldb + kn, 32u, 128u,
                       (unsigned long long)ldb);
        else
          tdm_issue_2d((unsigned)(size_t)&BbufS[nb][0],
                       Bm + (size_t)kn * ldb + bn, 128u, 32u,
                       (unsigned long long)ldb);
        __builtin_amdgcn_s_wait_tensorcnt((short)2);  // current pair done
      } else {
        __builtin_amdgcn_s_wait_tensorcnt((short)0);
      }
    }
    __syncthreads();
#else
    const int cur = 0;
    __syncthreads();  // previous compute finished before overwrite
    // ---- stage A tile 128(M) x 32(K) bf16: 512 x 16B, 2 per thread ----
#pragma unroll
    for (int s2 = 0; s2 < 2; ++s2) {
      int lin = tid + s2 * 256;
      int row = lin >> 2;
      int col = (lin & 3) << 3;
      *reinterpret_cast<uint4*>(&AbufS[0][row * 32 + col]) =
          *reinterpret_cast<const uint4*>(A + (size_t)(bm + row) * lda + kt + col);
    }
    if (!TRANSB) {
#pragma unroll
      for (int s2 = 0; s2 < 2; ++s2) {
        int lin = tid + s2 * 256;
        int row = lin >> 2;
        int col = (lin & 3) << 3;
        *reinterpret_cast<uint4*>(&BbufS[0][row * 32 + col]) =
            *reinterpret_cast<const uint4*>(Bm + (size_t)(bn + row) * ldb + kt + col);
      }
    } else {
#pragma unroll
      for (int s2 = 0; s2 < 2; ++s2) {
        int lin = tid + s2 * 256;
        int kk  = lin >> 4;
        int x   = (lin & 15) << 3;
        *reinterpret_cast<uint4*>(&BbufS[0][kk * 128 + x]) =
            *reinterpret_cast<const uint4*>(Bm + (size_t)(kt + kk) * ldb + bn + x);
      }
    }
    if (kt + 32 < K) {
      __builtin_prefetch(A + (size_t)(bm + (tid >> 1)) * lda + (kt + 32), 0, 1);
      __builtin_prefetch(Bm + (TRANSB ? ((size_t)(kt + 32 + (tid & 31)) * ldb + bn)
                                      : ((size_t)(bn + (tid >> 1)) * ldb + kt + 32)),
                         0, 1);
    }
    __syncthreads();
#endif

    // ---- fragments from LDS ----
    const unsigned short* Ab = &AbufS[cur][0];
    const unsigned short* Bb = &BbufS[cur][0];

    FragAB a[4];
#pragma unroll
    for (int i = 0; i < 4; ++i) {
      int row = wm * 64 + i * 16 + l16;
      a[i].h[0] = *reinterpret_cast<const bf16x8*>(Ab + row * 32 + half * 16);
      a[i].h[1] = *reinterpret_cast<const bf16x8*>(Ab + row * 32 + half * 16 + 8);
    }
    bf16x16 bfr[2];
#pragma unroll
    for (int j = 0; j < 2; ++j) {
      int n = wn * 32 + j * 16 + l16;
      if (!TRANSB) {
        bfr[j] = *reinterpret_cast<const bf16x16*>(Bb + n * 32 + half * 16);
      } else {
        FragAB u;
        const __bf16* bp = reinterpret_cast<const __bf16*>(Bb);
#pragma unroll
        for (int e = 0; e < 16; ++e) u.e[e] = bp[(half * 16 + e) * 128 + n];
        bfr[j] = u.v;
      }
    }

#pragma unroll
    for (int i = 0; i < 4; ++i)
#pragma unroll
      for (int j = 0; j < 2; ++j)
        acc[i][j] = __builtin_amdgcn_wmma_f32_16x16x32_bf16(
            false, a[i].v, false, bfr[j], (short)0, acc[i][j], false, false);

#if USE_TDM
    __syncthreads();  // this buffer free before it+2's DMA targets it
#endif
  }

  // ---- epilogue ----
#pragma unroll
  for (int j = 0; j < 2; ++j) {
    const int cg = bn + wn * 32 + j * 16 + l16;
    float bval = 0.0f;
    if (MODE == 0) bval = bias[cg];
#pragma unroll
    for (int i = 0; i < 4; ++i) {
#pragma unroll
      for (int r = 0; r < 8; ++r) {
        const int rg = bm + wm * 64 + i * 16 + half * 8 + r;
        float v = acc[i][j][r] * alpha + bval;
        if (MODE == 1) {
          bool msk = (rg < m_len) ? (cg >= m_len)
                                  : ((cg >= m_len) && (cg > rg));
          if (msk) v = -__builtin_inff();
        }
        if (MODE != 2) C[(size_t)rg * ldc + cg] = v;
        if ((MODE == 2) || (MODE == 0 && C2)) C2[(size_t)rg * ldc + cg] = f2bf(v);
      }
    }
  }
}

// ---------------------------------------------------------------------------
// Streaming f32 -> bf16 conversion (n multiple of 4).
// ---------------------------------------------------------------------------
__global__ __launch_bounds__(256) void f32_to_bf16_kernel(
    const float* __restrict__ src, unsigned short* __restrict__ dst, long n)
{
  long i = ((long)blockIdx.x * 256 + threadIdx.x) * 4;
  if (i + 3 < n) {
    float4 f = *reinterpret_cast<const float4*>(src + i);
    *reinterpret_cast<unsigned long long*>(dst + i) = pack4bf(f);
  }
}

// ---------------------------------------------------------------------------
// RoPE: q raw f32 (ws) -> roped bf16 qb; k raw f32 (cache region) -> roped
// f32 in place (required output) + roped bf16 kb.  One (d, d+64) pair/thread.
// ---------------------------------------------------------------------------
__global__ __launch_bounds__(256) void rope_kernel(
    const float* __restrict__ qf, float* __restrict__ kf,
    unsigned short* __restrict__ qb, unsigned short* __restrict__ kb)
{
  const int S = 2048, H = 16;
  const int idx = blockIdx.x * 256 + threadIdx.x;   // B*S*H*64 threads exactly
  const int d = idx & 63;
  const int h = (idx >> 6) & 15;
  const int s = (idx >> 10) & 2047;
  const int b = idx >> 21;

  const float ang = (float)s * __expf(-(float)d * (9.210340371976184f / 64.0f));
  float c, sn;
  __sincosf(ang, &sn, &c);

  const size_t base = (((size_t)b * S + s) * H + h) * 128 + d;
  float x1 = qf[base], x2 = qf[base + 64];
  qb[base]      = f2bf(x1 * c - x2 * sn);
  qb[base + 64] = f2bf(x2 * c + x1 * sn);
  float y1 = kf[base], y2 = kf[base + 64];
  float k1 = y1 * c - y2 * sn, k2 = y2 * c + y1 * sn;
  kf[base] = k1;          kf[base + 64] = k2;
  kb[base] = f2bf(k1);    kb[base + 64] = f2bf(k2);
}

// ---------------------------------------------------------------------------
// Row softmax over attn_w in place (f32 output) + bf16 copy for the AV GEMM.
// ---------------------------------------------------------------------------
__global__ __launch_bounds__(256) void softmax_kernel(
    float* __restrict__ attn, unsigned short* __restrict__ attnb, int T)
{
  __shared__ float red[8];
  const size_t base = (size_t)blockIdx.x * T;
  const int tid = threadIdx.x;

  float lmax = -3.4e38f;
  for (int t = tid; t < T; t += 256) lmax = fmaxf(lmax, attn[base + t]);
#pragma unroll
  for (int off = 16; off > 0; off >>= 1)
    lmax = fmaxf(lmax, __shfl_xor(lmax, off, 32));
  if ((tid & 31) == 0) red[tid >> 5] = lmax;
  __syncthreads();
  float rmax = red[0];
#pragma unroll
  for (int w = 1; w < 8; ++w) rmax = fmaxf(rmax, red[w]);
  __syncthreads();

  float lsum = 0.0f;
  for (int t = tid; t < T; t += 256) lsum += __expf(attn[base + t] - rmax);
#pragma unroll
  for (int off = 16; off > 0; off >>= 1) lsum += __shfl_xor(lsum, off, 32);
  if ((tid & 31) == 0) red[tid >> 5] = lsum;
  __syncthreads();
  float rsum = 0.0f;
#pragma unroll
  for (int w = 0; w < 8; ++w) rsum += red[w];
  const float inv = 1.0f / rsum;

  for (int t = tid; t < T; t += 256) {
    float v = __expf(attn[base + t] - rmax) * inv;
    attn[base + t]  = v;
    attnb[base + t] = f2bf(v);
  }
}

// ---------------------------------------------------------------------------
// Host launcher.
// Inputs: 0 x, 1 memory_length, 2 cache_k(empty), 3 cache_v(empty),
//         4 Wq, 5 bq, 6 Wk, 7 bk, 8 Wv, 9 bv, 10 Wo, 11 bo
// d_out:  out [B,S,HID] | attn_w [B,H,S,T] | cache_k [B,T,H,D] | cache_v
// ---------------------------------------------------------------------------
extern "C" void kernel_launch(void* const* d_in, const int* in_sizes, int n_in,
                              void* d_out, int out_size, void* d_ws, size_t ws_size,
                              hipStream_t stream)
{
  const float* x    = (const float*)d_in[0];
  const int*   mptr = (const int*)d_in[1];
  const float* Wq   = (const float*)d_in[4];
  const float* bq   = (const float*)d_in[5];
  const float* Wk   = (const float*)d_in[6];
  const float* bk   = (const float*)d_in[7];
  const float* Wv   = (const float*)d_in[8];
  const float* bv   = (const float*)d_in[9];
  const float* Wo   = (const float*)d_in[10];
  const float* bo   = (const float*)d_in[11];

  const long B = 2, S = 2048, HID = 2048, H = 16, D = 128, T = 2048;
  const long M = B * S;

  float* out  = (float*)d_out;
  float* attn = out  + (size_t)B * S * HID;          // [B,H,S,T]
  float* ck   = attn + (size_t)B * H * S * T;        // cache_k [B,T,H,D]
  float* cv   = ck   + (size_t)B * S * HID;          // cache_v

  // workspace partition
  char* w = (char*)d_ws;
  unsigned short* xb    = (unsigned short*)w; w += (size_t)M * HID * 2;
  unsigned short* Wqb   = (unsigned short*)w; w += (size_t)HID * HID * 2;
  unsigned short* Wkb   = (unsigned short*)w; w += (size_t)HID * HID * 2;
  unsigned short* Wvb   = (unsigned short*)w; w += (size_t)HID * HID * 2;
  unsigned short* Wob   = (unsigned short*)w; w += (size_t)HID * HID * 2;
  float*          qf    = (float*)w;          w += (size_t)M * HID * 4;
  unsigned short* qb    = (unsigned short*)w; w += (size_t)M * HID * 2;
  unsigned short* kb    = (unsigned short*)w; w += (size_t)M * HID * 2;
  unsigned short* vb    = (unsigned short*)w; w += (size_t)M * HID * 2;
  unsigned short* attnb = (unsigned short*)w; w += (size_t)B * H * S * T * 2;
  unsigned short* ctxb  = (unsigned short*)w;

  dim3 blk(256);
  const float iscale = 0.08838834764831845f;          // 1/sqrt(128)

  // --- pre-convert GEMM operands to bf16 (one streaming pass each) ---
  f32_to_bf16_kernel<<<(unsigned)(M * HID / 4 / 256), blk, 0, stream>>>(x, xb, M * HID);
  f32_to_bf16_kernel<<<(unsigned)(HID * HID / 4 / 256), blk, 0, stream>>>(Wq, Wqb, HID * HID);
  f32_to_bf16_kernel<<<(unsigned)(HID * HID / 4 / 256), blk, 0, stream>>>(Wk, Wkb, HID * HID);
  f32_to_bf16_kernel<<<(unsigned)(HID * HID / 4 / 256), blk, 0, stream>>>(Wv, Wvb, HID * HID);
  f32_to_bf16_kernel<<<(unsigned)(HID * HID / 4 / 256), blk, 0, stream>>>(Wo, Wob, HID * HID);

  // --- QKV projections (K/V f32 straight into cache output slots;
  //     V also emits bf16 for the AV GEMM) ---
  wmma_gemm_kernel<0, false><<<dim3(M / 128, HID / 128, 1), blk, 0, stream>>>(
      xb, HID, 0, 0, Wqb, HID, 0, 0, bq, qf, HID, 0, 0,
      nullptr, 0, 0, (int)HID, 1, 1.0f, nullptr);
  wmma_gemm_kernel<0, false><<<dim3(M / 128, HID / 128, 1), blk, 0, stream>>>(
      xb, HID, 0, 0, Wkb, HID, 0, 0, bk, ck, HID, 0, 0,
      nullptr, 0, 0, (int)HID, 1, 1.0f, nullptr);
  wmma_gemm_kernel<0, false><<<dim3(M / 128, HID / 128, 1), blk, 0, stream>>>(
      xb, HID, 0, 0, Wvb, HID, 0, 0, bv, cv, HID, 0, 0,
      vb, 0, 0, (int)HID, 1, 1.0f, nullptr);

  // --- RoPE ---
  rope_kernel<<<(unsigned)(B * S * H * 64 / 256), blk, 0, stream>>>(qf, ck, qb, kb);

  // --- scores = q . k^T / sqrt(D), masked -> attn_w region (f32) ---
  wmma_gemm_kernel<1, false><<<dim3(S / 128, T / 128, B * H), blk, 0, stream>>>(
      qb, HID, S * HID, D,
      kb, HID, S * HID, D,
      nullptr,
      attn, T, H * S * T, S * T,
      nullptr, 0, 0,
      (int)D, (int)H, iscale, mptr);

  // --- softmax rows in place + bf16 copy ---
  softmax_kernel<<<(unsigned)(B * H * S), blk, 0, stream>>>(attn, attnb, (int)T);

  // --- context = attn @ v  (v is [K=T, N=D] bf16 -> TRANSB), bf16 out ---
  wmma_gemm_kernel<2, true><<<dim3(S / 128, D / 128, B * H), blk, 0, stream>>>(
      attnb, T, H * S * T, S * T,
      vb,    HID, S * HID, D,
      nullptr,
      nullptr, HID, 0, 0,
      ctxb, S * HID, D,
      (int)T, (int)H, 1.0f, nullptr);

  // --- output projection ---
  wmma_gemm_kernel<0, false><<<dim3(M / 128, HID / 128, 1), blk, 0, stream>>>(
      ctxb, HID, 0, 0, Wob, HID, 0, 0, bo, out, HID, 0, 0,
      nullptr, 0, 0, (int)HID, 1, 1.0f, nullptr);
}
